// TriangleAttention_30760555774407
// MI455X (gfx1250) — compile-verified, matrix operand order checked by reading, not run on hardware
//
#include <hip/hip_runtime.h>
#include <hip/hip_bf16.h>

typedef unsigned short u16;
typedef __attribute__((ext_vector_type(8)))  float  v8f;
typedef __attribute__((ext_vector_type(16))) __bf16 bf16x16;
typedef __attribute__((ext_vector_type(4)))  unsigned int u32x4;
typedef __attribute__((ext_vector_type(4)))  int i32x4;
typedef __attribute__((ext_vector_type(8)))  int i32x8;

union Frag {
    uint4   q[2];   // 32 bytes
    bf16x16 b;
};

__device__ __forceinline__ u16 f2bf(float f) {
    unsigned int u = __float_as_uint(f);
    unsigned int r = u + 0x7FFFu + ((u >> 16) & 1u);
    return (u16)(r >> 16);
}
__device__ __forceinline__ float bf2f(u16 h) {
    return __uint_as_float(((unsigned int)h) << 16);
}

// ---------------------------------------------------------------------------
// Kernel 1: weight prep.  PW[512][128] bf16 (rows = output cols of wq|wk|wv|wg,
// contiguous along K=C).  OW[128][128] bf16 (row c = column c of wo).
// ---------------------------------------------------------------------------
__global__ __launch_bounds__(128) void prep_weights(
    const float* __restrict__ wq, const float* __restrict__ wk,
    const float* __restrict__ wv, const float* __restrict__ wg,
    const float* __restrict__ wo, u16* __restrict__ PW, u16* __restrict__ OW)
{
    int r = blockIdx.x;        // 0..639
    int c = threadIdx.x;       // 0..127  (K index)
    if (r < 512) {
        int sel = r >> 7, n = r & 127;
        const float* W = (sel == 0) ? wq : (sel == 1) ? wk : (sel == 2) ? wv : wg;
        PW[r * 128 + c] = f2bf(W[c * 128 + n]);      // W[c][n] -> row n
    } else {
        int cc = r - 512;                            // output column of final GEMM
        OW[cc * 128 + c] = f2bf(wo[c * 128 + cc]);   // wo[d][cc] -> row cc
    }
}

// ---------------------------------------------------------------------------
// Kernel 2: LayerNorm (one wave per 128-elem row) + triangle bias (H=4 dots).
// xln: [102400][128] bf16.  tri: [H][320][320] f32.
// ---------------------------------------------------------------------------
__global__ __launch_bounds__(256) void ln_kernel(
    const float* __restrict__ x, const float* __restrict__ g,
    const float* __restrict__ b, const float* __restrict__ wb,
    u16* __restrict__ xln, float* __restrict__ tri)
{
    int t = threadIdx.x, lane = t & 31, w = t >> 5;
    size_t m = (size_t)blockIdx.x * 8 + w;           // row index 0..102399
    const float* xr = x + m * 128;
    int c0 = lane * 4;
    float4 xv = *(const float4*)(xr + c0);
    float xa[4] = { xv.x, xv.y, xv.z, xv.w };

    float s = xa[0] + xa[1] + xa[2] + xa[3];
    float s2 = xa[0]*xa[0] + xa[1]*xa[1] + xa[2]*xa[2] + xa[3]*xa[3];
    #pragma unroll
    for (int d = 16; d >= 1; d >>= 1) {
        s  += __shfl_xor(s,  d, 32);
        s2 += __shfl_xor(s2, d, 32);
    }
    float mu  = s * (1.0f / 128.0f);
    float var = s2 * (1.0f / 128.0f) - mu * mu;
    float rs  = rsqrtf(var + 1e-5f);

    float y[4];
    u16   ys[4];
    #pragma unroll
    for (int e = 0; e < 4; ++e) {
        y[e]  = (xa[e] - mu) * rs * g[c0 + e] + b[c0 + e];
        ys[e] = f2bf(y[e]);
    }
    uint2 pk;
    pk.x = (unsigned)ys[0] | ((unsigned)ys[1] << 16);
    pk.y = (unsigned)ys[2] | ((unsigned)ys[3] << 16);
    *(uint2*)(xln + m * 128 + c0) = pk;

    // triangle bias: tri[h][i][j] = dot(y_row, wb[:,h])
    float th[4];
    #pragma unroll
    for (int h = 0; h < 4; ++h) {
        th[h] = y[0]*wb[(c0+0)*4 + h] + y[1]*wb[(c0+1)*4 + h]
              + y[2]*wb[(c0+2)*4 + h] + y[3]*wb[(c0+3)*4 + h];
        #pragma unroll
        for (int d = 16; d >= 1; d >>= 1) th[h] += __shfl_xor(th[h], d, 32);
    }
    if (lane == 0) {
        size_t i = m / 320, j = m % 320;
        #pragma unroll
        for (int h = 0; h < 4; ++h)
            tri[((size_t)h * 320 + i) * 320 + j] = th[h];
    }
}

// ---------------------------------------------------------------------------
// LDS staging + 128x128x128 bf16 WMMA compute helpers.
// 256 threads = 8 waves in 2(M)x4(N); each wave 4x2 tiles of
// v_wmma_f32_16x16x32_bf16.  Rows padded to 136 halves (conflict-free).
// ---------------------------------------------------------------------------
__device__ __forceinline__ void load_tile128(
    const u16* __restrict__ src, size_t row0, u16* dstLDS)
{
    int t = threadIdx.x;
    #pragma unroll
    for (int r = 0; r < 8; ++r) {
        int idx = r * 256 + t;                // 0..2047 uint4s
        int row = idx >> 4;
        int seg = (idx & 15) << 3;            // halves
        *(uint4*)&dstLDS[row * 136 + seg] =
            *(const uint4*)(src + (row0 + row) * 128 + seg);
    }
}

__device__ __forceinline__ void gemm128_compute(
    const u16* As, const u16* Bs, v8f acc[4][2])
{
    int t = threadIdx.x;
    int lane = t & 31, w = t >> 5, wm = w >> 2, wn = w & 3;
    int half = (lane < 16) ? 0 : 8;
    int lr   = lane & 15;

    #pragma unroll
    for (int kk = 0; kk < 4; ++kk) {
        Frag a[4], bfr[2];
        #pragma unroll
        for (int mt = 0; mt < 4; ++mt) {
            const u16* p = &As[(wm * 64 + mt * 16 + lr) * 136 + kk * 32 + half];
            a[mt].q[0] = *(const uint4*)p;
            a[mt].q[1] = *(const uint4*)(p + 16);
        }
        #pragma unroll
        for (int nt = 0; nt < 2; ++nt) {
            const u16* p = &Bs[(wn * 32 + nt * 16 + lr) * 136 + kk * 32 + half];
            bfr[nt].q[0] = *(const uint4*)p;
            bfr[nt].q[1] = *(const uint4*)(p + 16);
        }
        #pragma unroll
        for (int mt = 0; mt < 4; ++mt)
            #pragma unroll
            for (int nt = 0; nt < 2; ++nt)
                acc[mt][nt] = __builtin_amdgcn_wmma_f32_16x16x32_bf16(
                    false, a[mt].b, false, bfr[nt].b, (short)0, acc[mt][nt],
                    false, false);
    }
}

// ---------------------------------------------------------------------------
// Kernel 3: projection GEMM.  A-tile loaded ONCE, then loop over the four
// weight tiles (q/k/v/g) -> 4x less x_ln traffic.
// Output layout: [(i*4+h)*320 + j][dd] bf16 (attention friendly).
// ---------------------------------------------------------------------------
__global__ __launch_bounds__(256) void proj_gemm(
    const u16* __restrict__ xln, const u16* __restrict__ PW,
    const float* __restrict__ bg, u16* __restrict__ qb, u16* __restrict__ kb,
    u16* __restrict__ vb, u16* __restrict__ gb)
{
    __shared__ __align__(16) u16 As[128 * 136];
    __shared__ __align__(16) u16 Bs[128 * 136];

    size_t m0 = (size_t)blockIdx.x * 128;
    load_tile128(xln, m0, As);

    int t = threadIdx.x, lane = t & 31, w = t >> 5, wm = w >> 2, wn = w & 3;
    int half = (lane < 16) ? 0 : 8, lr = lane & 15;
    u16* dsts[4] = { qb, kb, vb, gb };

    for (int sel = 0; sel < 4; ++sel) {
        if (sel) __syncthreads();             // WAR on Bs vs previous compute
        load_tile128(PW + (size_t)sel * 128 * 128, 0, Bs);
        __syncthreads();

        v8f acc[4][2];
        #pragma unroll
        for (int mt = 0; mt < 4; ++mt)
            #pragma unroll
            for (int nt = 0; nt < 2; ++nt)
                #pragma unroll
                for (int e = 0; e < 8; ++e) acc[mt][nt][e] = 0.0f;

        gemm128_compute(As, Bs, acc);

        u16* dst = dsts[sel];
        float scale = (sel == 0) ? 0.17677669529663689f : 1.0f;   // 1/sqrt(32)
        #pragma unroll
        for (int mt = 0; mt < 4; ++mt)
            #pragma unroll
            for (int nt = 0; nt < 2; ++nt)
                #pragma unroll
                for (int vv = 0; vv < 8; ++vv) {
                    int col = wn * 32 + nt * 16 + lr;             // 0..127
                    size_t m = m0 + wm * 64 + mt * 16 + half + vv;
                    size_t i = m / 320, j = m % 320;
                    int hh = col >> 5, dd = col & 31;
                    float val = acc[mt][nt][vv];
                    if (sel == 3) val = 1.0f / (1.0f + __expf(-(val + bg[col])));
                    else          val *= scale;
                    dst[((i * 4 + hh) * 320 + j) * 32 + dd] = f2bf(val);
                }
    }
}

// ---------------------------------------------------------------------------
// Kernel 4: attention.  One block per (i,h); K staged via TDM
// (tensor_load_to_lds with pad_enable producing the 40-half padded stride),
// V^T staged manually, then 5 query tiles of 64 processed in-block.
// ---------------------------------------------------------------------------
__global__ __launch_bounds__(128) void attn_kernel(
    const u16* __restrict__ qg, const u16* __restrict__ kg,
    const u16* __restrict__ vg, const u16* __restrict__ gg,
    const float* __restrict__ tri, const float* __restrict__ mask,
    u16* __restrict__ og)
{
    __shared__ __align__(16) u16 Ks[320 * 40];    // K rows padded to 40 halves
    __shared__ __align__(16) u16 Vt[32 * 328];    // V^T rows padded to 328
    __shared__ __align__(16) u16 Ps[64 * 328];    // probs, bf16 (per-wave rows)

    int t = threadIdx.x, lane = t & 31, w = t >> 5;
    int ih = blockIdx.x;                          // i*4 + h
    int i  = ih >> 2, h = ih & 3;
    int half = (lane < 16) ? 0 : 8;
    int lr   = lane & 15;

    const u16* Kg = kg + (size_t)ih * 320 * 32;
    const u16* Vg = vg + (size_t)ih * 320 * 32;

    // ---- TDM: DMA K [320 rows x 32 bf16] into Ks with 16B pad per 64B row
    if (w == 0) {
        unsigned           ldsK = (unsigned)(size_t)(void*)Ks;
        unsigned long long ga   = (unsigned long long)(size_t)(const void*)Kg;
        u32x4 g0;
        g0[0] = 1u;                                       // count=1, user mode
        g0[1] = ldsK;                                     // lds_addr
        g0[2] = (unsigned)(ga & 0xFFFFFFFFu);             // global_addr lo
        g0[3] = (unsigned)((ga >> 32) & 0x1FFFFFFu) | (2u << 30); // hi | type=2
        i32x8 g1;
        g1[0] = (int)((1u << 16)      // data_size = 2 bytes
                    | (1u << 20)      // pad_enable
                    | (3u << 22)      // pad_interval: 16 DWORDs (one 64B row)
                    | (3u << 25));    // pad_amount: 4 DWORDs (16B) -> 80B rows
        g1[1] = (int)(32u  << 16);    // tensor_dim0 = 32  (bits 79:48, lo16)
        g1[2] = (int)(320u << 16);    // tensor_dim1 = 320 (bits 111:80, lo16)
        g1[3] = (int)(32u  << 16);    // tile_dim0 = 32    (bits 127:112)
        g1[4] = 320;                  // tile_dim1 = 320
        g1[5] = 32;                   // tensor_dim0_stride = 32 elements
        g1[6] = 0;
        g1[7] = 0;
        i32x4 z4;
        z4[0] = 0; z4[1] = 0; z4[2] = 0; z4[3] = 0;
        i32x8 z8;
        #pragma unroll
        for (int e = 0; e < 8; ++e) z8[e] = 0;
        __builtin_amdgcn_tensor_load_to_lds(g0, g1, z4, z4, z8, 0);
        __builtin_amdgcn_s_wait_tensorcnt((short)0);
    }

    // ---- stage V^T manually (TDM cannot transpose)
    #pragma unroll
    for (int r = 0; r < 80; ++r) {
        int idx = r * 128 + t;                    // 0..10239 elems
        int row = idx >> 5, d = idx & 31;
        Vt[d * 328 + row] = Vg[row * 32 + d];
    }
    __syncthreads();

    // ---- loop over 5 query tiles of 64 (16 rows per wave)
    for (int qt = 0; qt < 5; ++qt) {
        int q0 = qt * 64 + w * 16;

        Frag qa;
        {
            const u16* Qr = qg + ((size_t)ih * 320 + q0 + lr) * 32 + half;
            qa.q[0] = *(const uint4*)Qr;
            qa.q[1] = *(const uint4*)(Qr + 16);
        }

        // scores: 20 key tiles of 16, add mask + triangle bias per element
        v8f sc[20];
        #pragma unroll
        for (int kt = 0; kt < 20; ++kt) {
            Frag bk;
            const u16* kp = &Ks[(kt * 16 + lr) * 40 + half];
            bk.q[0] = *(const uint4*)kp;
            bk.q[1] = *(const uint4*)(kp + 16);
            v8f c;
            #pragma unroll
            for (int e = 0; e < 8; ++e) c[e] = 0.0f;
            c = __builtin_amdgcn_wmma_f32_16x16x32_bf16(
                false, qa.b, false, bk.b, (short)0, c, false, false);
            int col = kt * 16 + lr;
            float mb = 1.0e9f * (mask[(size_t)i * 320 + col] - 1.0f);
            const float* tp = tri + ((size_t)h * 320 + q0 + half) * 320 + col;
            #pragma unroll
            for (int vv = 0; vv < 8; ++vv)
                c[vv] += mb + tp[(size_t)vv * 320];
            sc[kt] = c;
        }

        // softmax per row (rows live in one 16-lane half; reduce xor 8,4,2,1)
        #pragma unroll
        for (int vv = 0; vv < 8; ++vv) {
            float mx = -1.0e30f;
            #pragma unroll
            for (int kt = 0; kt < 20; ++kt) mx = fmaxf(mx, sc[kt][vv]);
            #pragma unroll
            for (int d = 8; d >= 1; d >>= 1) mx = fmaxf(mx, __shfl_xor(mx, d, 32));
            float s = 0.0f;
            #pragma unroll
            for (int kt = 0; kt < 20; ++kt) {
                float e = __expf(sc[kt][vv] - mx);
                sc[kt][vv] = e;
                s += e;
            }
            #pragma unroll
            for (int d = 8; d >= 1; d >>= 1) s += __shfl_xor(s, d, 32);
            float inv = 1.0f / s;
            int prow = w * 16 + half + vv;
            #pragma unroll
            for (int kt = 0; kt < 20; ++kt)
                Ps[prow * 328 + kt * 16 + lr] = f2bf(sc[kt][vv] * inv);
        }

        // O = P @ V  (K = 320, 10 chunks of 32; CH=32 -> 2 output tiles)
        v8f O[2];
        #pragma unroll
        for (int dt = 0; dt < 2; ++dt)
            #pragma unroll
            for (int e = 0; e < 8; ++e) O[dt][e] = 0.0f;

        #pragma unroll
        for (int kk = 0; kk < 10; ++kk) {
            Frag pa;
            const u16* pp = &Ps[(w * 16 + lr) * 328 + kk * 32 + half];
            pa.q[0] = *(const uint4*)pp;
            pa.q[1] = *(const uint4*)(pp + 16);
            #pragma unroll
            for (int dt = 0; dt < 2; ++dt) {
                Frag vb;
                const u16* vp = &Vt[(dt * 16 + lr) * 328 + kk * 32 + half];
                vb.q[0] = *(const uint4*)vp;
                vb.q[1] = *(const uint4*)(vp + 16);
                O[dt] = __builtin_amdgcn_wmma_f32_16x16x32_bf16(
                    false, pa.b, false, vb.b, (short)0, O[dt], false, false);
            }
        }

        // gate and store (layout [i][j][h*32+d] bf16 for the output GEMM)
        #pragma unroll
        for (int dt = 0; dt < 2; ++dt)
            #pragma unroll
            for (int vv = 0; vv < 8; ++vv) {
                int qrow = q0 + half + vv;
                int d = dt * 16 + lr;
                float gate = bf2f(gg[((size_t)ih * 320 + qrow) * 32 + d]);
                float val = O[dt][vv] * gate;
                og[((size_t)i * 320 + qrow) * 128 + h * 32 + d] = f2bf(val);
            }
    }
}

// ---------------------------------------------------------------------------
// Kernel 5: output GEMM: out = og @ wo + bo  (f32 out)
// ---------------------------------------------------------------------------
__global__ __launch_bounds__(256) void out_gemm(
    const u16* __restrict__ og, const u16* __restrict__ OW,
    const float* __restrict__ bo, float* __restrict__ out)
{
    __shared__ __align__(16) u16 As[128 * 136];
    __shared__ __align__(16) u16 Bs[128 * 136];

    size_t m0 = (size_t)blockIdx.x * 128;
    load_tile128(og, m0, As);
    load_tile128(OW, 0, Bs);
    __syncthreads();

    v8f acc[4][2];
    #pragma unroll
    for (int mt = 0; mt < 4; ++mt)
        #pragma unroll
        for (int nt = 0; nt < 2; ++nt)
            #pragma unroll
            for (int e = 0; e < 8; ++e) acc[mt][nt][e] = 0.0f;

    gemm128_compute(As, Bs, acc);

    int t = threadIdx.x, lane = t & 31, w = t >> 5, wm = w >> 2, wn = w & 3;
    int half = (lane < 16) ? 0 : 8, lr = lane & 15;

    #pragma unroll
    for (int mt = 0; mt < 4; ++mt)
        #pragma unroll
        for (int nt = 0; nt < 2; ++nt)
            #pragma unroll
            for (int vv = 0; vv < 8; ++vv) {
                int col = wn * 32 + nt * 16 + lr;
                size_t m = m0 + wm * 64 + mt * 16 + half + vv;
                out[m * 128 + col] = acc[mt][nt][vv] + bo[col];
            }
}

// ---------------------------------------------------------------------------
extern "C" void kernel_launch(void* const* d_in, const int* in_sizes, int n_in,
                              void* d_out, int out_size, void* d_ws, size_t ws_size,
                              hipStream_t stream)
{
    const float* x     = (const float*)d_in[0];
    const float* mask  = (const float*)d_in[1];
    const float* ln_g  = (const float*)d_in[2];
    const float* ln_b  = (const float*)d_in[3];
    const float* wbias = (const float*)d_in[4];
    const float* wq    = (const float*)d_in[5];
    const float* wk    = (const float*)d_in[6];
    const float* wv    = (const float*)d_in[7];
    const float* wg    = (const float*)d_in[8];
    const float* bg    = (const float*)d_in[9];
    const float* wo    = (const float*)d_in[10];
    const float* bo    = (const float*)d_in[11];

    const size_t MROWS = 320 * 320;               // 102400
    char* ws = (char*)d_ws;
    u16*   xln = (u16*)ws;   ws += MROWS * 128 * 2;   // 26.2 MB
    u16*   PW  = (u16*)ws;   ws += 512 * 128 * 2;
    u16*   OW  = (u16*)ws;   ws += 128 * 128 * 2;
    float* tri = (float*)ws; ws += 4 * 320 * 320 * 4; // 1.6 MB
    u16*   qb  = (u16*)ws;   ws += MROWS * 128 * 2;
    u16*   kb  = (u16*)ws;   ws += MROWS * 128 * 2;
    u16*   vb  = (u16*)ws;   ws += MROWS * 128 * 2;
    u16*   gb  = (u16*)ws;   ws += MROWS * 128 * 2;
    u16*   og  = (u16*)ws;   ws += MROWS * 128 * 2;

    prep_weights<<<640, 128, 0, stream>>>(wq, wk, wv, wg, wo, PW, OW);
    ln_kernel<<<12800, 256, 0, stream>>>(x, ln_g, ln_b, wbias, xln, tri);
    proj_gemm<<<dim3(800, 1), 256, 0, stream>>>(xln, PW, bg, qb, kb, vb, gb);
    attn_kernel<<<1280, 128, 0, stream>>>(qb, kb, vb, gb, tri, mask, og);
    out_gemm<<<800, 256, 0, stream>>>(og, OW, bo, (float*)d_out);
}
